// HyperDimensionalLoss_30751965839942
// MI455X (gfx1250) — matrix operand order repaired and checked by reading, not compiled, not run on hardware
//
#include <hip/hip_runtime.h>
#include <hip/hip_bf16.h>
#include <math.h>

typedef __attribute__((ext_vector_type(16))) _Float16 v16h;
typedef __attribute__((ext_vector_type(8)))  _Float16 v8h;
typedef __attribute__((ext_vector_type(8)))  float    v8f;

#define WINDOW   2048
#define STEP     256
#define NCOEF    1025
#define NB       4
#define NSAMP    131072
#define TFRM     506
#define TPAD     512
#define HDIM     16384

// gemm2 (projection) padded K: 33*32
#define KPAD2    1056
#define LDSROW   1064          // LDS row stride in halves (conflict-free for b128)
// gemm1 (DFT) pads
#define NPAD1    2176          // 17*128 >= 2*1025 Re/Im columns
#define MPAD1    2032          // 127*16 >= 4*506 frame rows
#define KC1      64            // 2048/32 k-chunks
#define KC2      33            // 1056/32 k-chunks

#define TWO_PI_OVER_W 0.0030679615757712823f   // 2*pi/2048
#define ORTHO_SCALE   0.022097086912079608f    // 1/sqrt(2048)

__device__ inline v8f wmma_f16(v16h a, v16h b, v8f c) {
    // D = A(16x32 f16) * B(32x16 f16) + C(16x16 f32)
    return __builtin_amdgcn_wmma_f32_16x16x32_f16(
        false, a, false, b, (short)0, c, false, false);
}

union ABFrag { v16h v; v8h h[2]; };

// ---------------------------------------------------------------------------
// k0: proj (1025 x 16384 f32, row-major) -> projB f16 in WMMA B-fragment order:
//     projB[((kc*16384 + n) * 32) + kk]  holds K = kc*32+kk, column n.
// Each thread transposes its own 32-element strip: coalesced reads (over n)
// and fully coalesced 64B writes. No barriers needed.
// ---------------------------------------------------------------------------
__global__ void k0_pack_proj(const float* __restrict__ proj,
                             _Float16* __restrict__ projB) {
    const int kc  = blockIdx.y;              // 0..32
    const int nb  = blockIdx.x;              // 0..63
    const int tid = threadIdx.x;             // 0..255
    const int n   = nb * 256 + tid;

    _Float16 strip[32];
#pragma unroll
    for (int r = 0; r < 32; ++r) {
        const int c = kc * 32 + r;
        float v = (c < NCOEF) ? proj[(size_t)c * HDIM + n] : 0.0f;
        strip[r] = (_Float16)v;
    }
    _Float16* dst = projB + ((size_t)(kc * HDIM + n) << 5);
#pragma unroll
    for (int j = 0; j < 4; ++j)
        *(v8h*)(dst + j * 8) = *(v8h*)(strip + j * 8);
}

// ---------------------------------------------------------------------------
// k1: DFT basis in B-fragment order.  Columns: n = 2c -> Re (cos), n = 2c+1
// -> Im (-sin);  n >= 2050 zero pad.  Twiddle via (c*k mod 2048) so sincos
// args stay in [0, 2pi) (exact range reduction, no large-angle loss).
// ---------------------------------------------------------------------------
__global__ void k1_gen_basis(_Float16* __restrict__ basisB) {
    const unsigned g = blockIdx.x * 256u + threadIdx.x;
    const unsigned total = (unsigned)KC1 * NPAD1 * 32u;
    if (g >= total) return;
    const unsigned kc = g / (NPAD1 * 32u);
    const unsigned rem = g - kc * (NPAD1 * 32u);
    const unsigned n  = rem >> 5;
    const unsigned kk = rem & 31u;
    const unsigned k  = kc * 32u + kk;

    float val = 0.0f;
    if (n < 2u * NCOEF) {
        const unsigned c = n >> 1;
        const unsigned m = (c * k) & (WINDOW - 1);   // ck mod 2048
        const float ang = (float)m * TWO_PI_OVER_W;
        float s, co;
        __sincosf(ang, &s, &co);
        val = (n & 1u) ? -s : co;                    // e^{-i 2pi ck/W}
    }
    basisB[g] = (_Float16)val;
}

// ---------------------------------------------------------------------------
// k2: windowed frames matrix f16, row-major K-contiguous (A-fragment source).
// row = b*506 + t (rows >= 2024 zeroed);  reference pads x by STEP zeros.
// ---------------------------------------------------------------------------
__global__ void k2_gen_frames(const float* __restrict__ x,
                              _Float16* __restrict__ frames) {
    const unsigned g = blockIdx.x * 256u + threadIdx.x;
    if (g >= (unsigned)MPAD1 * WINDOW) return;
    const unsigned row = g >> 11;
    const unsigned k   = g & (WINDOW - 1);
    float val = 0.0f;
    if (row < NB * TFRM) {
        const unsigned b = row / TFRM;
        const unsigned t = row - b * TFRM;
        const unsigned s = t * STEP + k;
        if (s < NSAMP) val = x[(size_t)b * NSAMP + s];
        const float win = 0.5f - 0.5f * __cosf((float)k * TWO_PI_OVER_W);
        val *= win;
    }
    frames[g] = (_Float16)val;
}

// ---------------------------------------------------------------------------
// k3: GEMM1 (DFT): reim[M=2032][N=2176] f32 = frames(2032x2048) x basis.
// 8 waves per WG spread over 128 columns; all waves share the same 16 A rows
// (tile re-reads hit the WGP cache; both operands are L2-resident).
// ---------------------------------------------------------------------------
__global__ void k3_gemm_dft(const _Float16* __restrict__ frames,
                            const _Float16* __restrict__ basisB,
                            float* __restrict__ reim) {
    const int wave = threadIdx.x >> 5;
    const int lane = threadIdx.x & 31;
    const int hi   = lane >> 4;                  // lane group 0/1
    const int row  = lane & 15;                  // A row within tile
    const int col  = blockIdx.x * 128 + wave * 16 + (lane & 15);
    const int m0   = blockIdx.y * 16;

    const _Float16* arow = frames + (size_t)(m0 + row) * WINDOW;
    v8f acc = {0.f, 0.f, 0.f, 0.f, 0.f, 0.f, 0.f, 0.f};

    for (int kc = 0; kc < KC1; ++kc) {
        ABFrag a, b;
        const int k0 = kc * 32 + hi * 16;        // {0..15} / {16..31}
        a.h[0] = *(const v8h*)(arow + k0);
        a.h[1] = *(const v8h*)(arow + k0 + 8);
        const _Float16* bp = basisB + (((size_t)kc * NPAD1 + col) << 5) + hi * 16;
        b.h[0] = *(const v8h*)bp;
        b.h[1] = *(const v8h*)(bp + 8);
        acc = wmma_f16(a.v, b.v, acc);
    }
#pragma unroll
    for (int i = 0; i < 8; ++i)                  // D: VGPR i -> M = i + 8*hi
        reim[(size_t)(m0 + hi * 8 + i) * NPAD1 + col] = acc[i];
}

// ---------------------------------------------------------------------------
// k4: |Re+iIm| * 1/sqrt(2048) -> specB f16 [4][512][1056], zero padded.
// ---------------------------------------------------------------------------
__global__ void k4_magnitude(const float* __restrict__ reim,
                             _Float16* __restrict__ specB) {
    const unsigned g = blockIdx.x * 256u + threadIdx.x;
    if (g >= (unsigned)NB * TPAD * KPAD2) return;
    const unsigned c   = g % KPAD2;
    const unsigned tmp = g / KPAD2;
    const unsigned t   = tmp & (TPAD - 1);
    const unsigned b   = tmp >> 9;
    float s = 0.0f;
    if (t < TFRM && c < NCOEF) {
        const size_t row = (size_t)b * TFRM + t;
        const float re = reim[row * NPAD1 + 2 * c];
        const float im = reim[row * NPAD1 + 2 * c + 1];
        s = sqrtf(re * re + im * im) * ORTHO_SCALE;
    }
    specB[g] = (_Float16)s;
}

// ---------------------------------------------------------------------------
// k5: GEMM2 (projection) + tanh + frame reduction, with double-buffered
// ASYNC global->LDS staging of the A tile (CDNA5 GLOBAL_LOAD_ASYNC_TO_LDS):
// the DMA for t-chunk tc+1 overlaps the 33 WMMAs of chunk tc; ASYNCcnt wait
// + workgroup barrier only at the chunk boundary.  2 x 34 KB LDS buffers
// (row stride 1064 halves -> conflict-free ds_load_b128 on consume).
// Each WG owns out[b, 128 cols] fully: tanh folded per accumulator lane into
// a running scalar, final shfl combine -> plain store, no atomics.
// ---------------------------------------------------------------------------
__global__ void k5_gemm_proj(const _Float16* __restrict__ specB,
                             const _Float16* __restrict__ projB,
                             float* __restrict__ out) {
    __shared__ _Float16 lds[2][16 * LDSROW];

    const int b    = blockIdx.y;
    const int wave = threadIdx.x >> 5;
    const int lane = threadIdx.x & 31;
    const int hi   = lane >> 4;
    const int row  = lane & 15;
    const int col  = blockIdx.x * 128 + wave * 16 + (lane & 15);

    // Issue async copies: 16 rows x 1056 halves of specB[b][tc*16 + r][*]
    // into lds[buf].  16B per lane per op, no VGPR write-back (ASYNCcnt).
    auto stage_async = [&](int tc, int buf) {
        for (int idx = threadIdx.x; idx < 16 * (KPAD2 / 8); idx += 256) {
            const int r  = idx / (KPAD2 / 8);
            const int c8 = idx - r * (KPAD2 / 8);
            const _Float16* gsrc =
                specB + ((size_t)(b * TPAD + tc * 16 + r) * KPAD2) + c8 * 8;
            const unsigned laddr =
                (unsigned)(uintptr_t)(&lds[buf][r * LDSROW + c8 * 8]);
            asm volatile("global_load_async_to_lds_b128 %0, %1, off"
                         :: "v"(laddr), "v"(gsrc) : "memory");
        }
    };

    stage_async(0, 0);                                  // prologue
    asm volatile("s_wait_asynccnt 0x0" ::: "memory");
    __syncthreads();

    float rsum = 0.0f;

    for (int tc = 0; tc < TPAD / 16; ++tc) {
        const int cur = tc & 1;
        if (tc + 1 < TPAD / 16)
            stage_async(tc + 1, cur ^ 1);               // overlaps WMMAs below

        v8f acc = {0.f, 0.f, 0.f, 0.f, 0.f, 0.f, 0.f, 0.f};
        for (int kc = 0; kc < KC2; ++kc) {
            ABFrag a, bf;
            const int k0 = kc * 32 + hi * 16;
            a.h[0] = *(const v8h*)(&lds[cur][row * LDSROW + k0]);
            a.h[1] = *(const v8h*)(&lds[cur][row * LDSROW + k0 + 8]);
            const _Float16* bp = projB + (((size_t)kc * HDIM + col) << 5) + hi * 16;
            bf.h[0] = *(const v8h*)bp;
            bf.h[1] = *(const v8h*)(bp + 8);
            if (kc + 1 < KC2)                           // global_prefetch next B tile
                __builtin_prefetch(bp + (size_t)HDIM * 32, 0, 3);
            acc = wmma_f16(a.v, bf.v, acc);
        }
#pragma unroll
        for (int i = 0; i < 8; ++i)                     // tanh + reduce over M rows
            rsum += tanhf(acc[i]);

        asm volatile("s_wait_asynccnt 0x0" ::: "memory");
        __syncthreads();                                // next tile visible to all waves
    }

    rsum += __shfl_down(rsum, 16, 32);                  // combine M 0-7 with 8-15
    if (lane < 16)
        out[(size_t)b * HDIM + col] = rsum;
}

// ---------------------------------------------------------------------------
extern "C" void kernel_launch(void* const* d_in, const int* in_sizes, int n_in,
                              void* d_out, int out_size, void* d_ws, size_t ws_size,
                              hipStream_t stream) {
    (void)in_sizes; (void)n_in; (void)out_size; (void)ws_size;
    const float* x    = (const float*)d_in[0];
    const float* proj = (const float*)d_in[1];
    float*       out  = (float*)d_out;

    char* ws = (char*)d_ws;
    size_t off = 0;
    auto take = [&](size_t bytes) { char* p = ws + off; off = (off + bytes + 255) & ~(size_t)255; return p; };

    _Float16* projB  = (_Float16*)take((size_t)KC2 * HDIM * 32 * 2);       // 34.6 MB
    _Float16* basisB = (_Float16*)take((size_t)KC1 * NPAD1 * 32 * 2);      //  8.9 MB
    _Float16* frames = (_Float16*)take((size_t)MPAD1 * WINDOW * 2);        //  8.3 MB
    float*    reim   = (float*)   take((size_t)MPAD1 * NPAD1 * 4);         // 17.7 MB
    _Float16* specB  = (_Float16*)take((size_t)NB * TPAD * KPAD2 * 2);     //  4.3 MB

    k0_pack_proj <<<dim3(HDIM / 256, KC2), 256, 0, stream>>>(proj, projB);

    { unsigned n = (unsigned)KC1 * NPAD1 * 32;
      k1_gen_basis <<<(n + 255) / 256, 256, 0, stream>>>(basisB); }

    { unsigned n = (unsigned)MPAD1 * WINDOW;
      k2_gen_frames<<<(n + 255) / 256, 256, 0, stream>>>(x, frames); }

    k3_gemm_dft  <<<dim3(NPAD1 / 128, MPAD1 / 16), 256, 0, stream>>>(frames, basisB, reim);

    { unsigned n = (unsigned)NB * TPAD * KPAD2;
      k4_magnitude <<<(n + 255) / 256, 256, 0, stream>>>(reim, specB); }

    k5_gemm_proj <<<dim3(HDIM / 128, NB), 256, 0, stream>>>(specB, projB, out);
}
